// TransR_22316650070810
// MI455X (gfx1250) — compile-verified, matrix operand order checked by reading, not run on hardware
//
#include <hip/hip_runtime.h>

typedef __attribute__((ext_vector_type(16))) _Float16 v16h;
typedef __attribute__((ext_vector_type(8)))  float    v8f;

#define DIM 128
#define GAMMA 12.0f
#define L2EPS 1e-12f

// ---------------------------------------------------------------------------
// Kernel 1: repack relation matrices f32 -> f16 in WMMA B-fragment-major order.
// Layout per relation rr (16384 halves = 32 KB):
//   frag f = e_tile*4 + k_chunk (32 frags), each frag = 32 lanes x 16 halves.
//   lane n (n = lane&15, hi = lane>>4) holds column e = e_tile*16 + n of the
//   B tile, K offsets {0-7,16-23} (lo lanes) / {8-15,24-31} (hi lanes),
//   matching the CDNA5 16-bit B-matrix VGPR layout for v_wmma_*_16x16x32.
// ---------------------------------------------------------------------------
__global__ void transr_repack_f16(const float* __restrict__ mat_w,
                                  _Float16* __restrict__ mat16,
                                  int total) {
    int idx = blockIdx.x * blockDim.x + threadIdx.x;
    if (idx >= total) return;
    int i    = idx & 15;          // half index within lane fragment
    int lane = (idx >> 4) & 31;   // lane within wave
    int f    = (idx >> 9) & 31;   // fragment: e_tile*4 + k_chunk
    int rr   = idx >> 14;         // relation id
    int et = f >> 2;
    int kc = f & 3;
    int n  = lane & 15;
    int hi = lane >> 4;
    int koff = (i < 8 ? i : i + 8) + hi * 8;
    int d = kc * 32 + koff;       // input (row) dimension of M
    int e = et * 16 + n;          // output (col) dimension of M
    mat16[idx] = (_Float16)mat_w[rr * (DIM * DIM) + d * DIM + e];
}

// ---------------------------------------------------------------------------
// Kernel 2: one wave32 per batch element. A rows: row0 = l2norm(hv),
// row1 = l2norm(tv), rows 2..15 zero. B = repacked relation matrix tiles.
// ---------------------------------------------------------------------------
__global__ void __launch_bounds__(256)
transr_score(const int* __restrict__ h_idx,
             const int* __restrict__ r_idx,
             const int* __restrict__ t_idx,
             const float* __restrict__ ent_w,
             const float* __restrict__ rel_w,
             const _Float16* __restrict__ mat16,
             float* __restrict__ out,
             int B, int n_rel) {
    const int lane = threadIdx.x & 31;
    const int wave = threadIdx.x >> 5;
    const int b = blockIdx.x * (blockDim.x >> 5) + wave;
    if (b >= B) return;  // wave-uniform

    const int h = h_idx[b];
    const int r = r_idx[b];
    const int t = t_idx[b];
    const int rr = (r < n_rel) ? r : r - n_rel;
    const float sign = (r < n_rel) ? 1.0f : -1.0f;

    const float* __restrict__ hv = ent_w + (long)h * DIM;
    const float* __restrict__ tv = ent_w + (long)t * DIM;
    const float* __restrict__ rv = rel_w + (long)rr * DIM;

    // ---- input norms: each lane owns 4 contiguous elements -------------
    float sh = 0.f, st = 0.f, sr = 0.f;
#pragma unroll
    for (int j = 0; j < 4; ++j) {
        float a = hv[lane * 4 + j]; sh += a * a;
        float c = tv[lane * 4 + j]; st += c * c;
        float d = rv[lane * 4 + j]; sr += d * d;
    }
#pragma unroll
    for (int m = 16; m >= 1; m >>= 1) {
        sh += __shfl_xor(sh, m, 32);
        st += __shfl_xor(st, m, 32);
        sr += __shfl_xor(sr, m, 32);
    }
    const float invH0 = 1.0f / fmaxf(sqrtf(sh), L2EPS);
    const float invT0 = 1.0f / fmaxf(sqrtf(st), L2EPS);
    const float invR  = sign / fmaxf(sqrtf(sr), L2EPS);

    // ---- build the 4 A fragments (K chunks of 32) ----------------------
    const int m  = lane & 15;  // A row this lane carries
    const int hi = lane >> 4;  // which K-half
    v16h afr[4];
#pragma unroll
    for (int kc = 0; kc < 4; ++kc) {
        v16h a = {};
        if (m < 2) {
            const float* src = (m == 0) ? hv : tv;
            const float  sc  = (m == 0) ? invH0 : invT0;
            const int base = kc * 32 + hi * 8;
#pragma unroll
            for (int i = 0; i < 16; ++i) {
                int koff = base + (i < 8 ? i : i + 8);
                a[i] = (_Float16)(src[koff] * sc);
            }
        }
        afr[kc] = a;
    }

    // ---- 8 output tiles x 4 K-chunks of v_wmma_f32_16x16x32_f16 --------
    const _Float16* __restrict__ mb = mat16 + (long)rr * (DIM * DIM);
    float hp8[8], tp8[8];
    float hss = 0.f, tss = 0.f;
#pragma unroll
    for (int et = 0; et < 8; ++et) {
        v8f c = {};
#pragma unroll
        for (int kc = 0; kc < 4; ++kc) {
            v16h bfr = *(const v16h*)(mb + (((et * 4 + kc) * 32 + lane) * 16));
            c = __builtin_amdgcn_wmma_f32_16x16x32_f16(
                    /*neg_a=*/false, afr[kc],
                    /*neg_b=*/false, bfr,
                    /*c_mod=*/(short)0, c,
                    /*reuse_a=*/false, /*reuse_b=*/false);
        }
        // C layout: VGPR0 lanes 0-15 = row M=0 (hp), VGPR1 lanes 0-15 = M=1 (tp).
        // Lanes 16-31 see rows 8/9 which are exact zeros.
        float hp = c[0];
        float tp = c[1];
        hp8[et] = hp;
        tp8[et] = tp;
        hss += hp * hp;
        tss += tp * tp;
    }
#pragma unroll
    for (int mm = 16; mm >= 1; mm >>= 1) {
        hss += __shfl_xor(hss, mm, 32);
        tss += __shfl_xor(tss, mm, 32);
    }
    const float invH = 1.0f / fmaxf(sqrtf(hss), L2EPS);
    const float invT = 1.0f / fmaxf(sqrtf(tss), L2EPS);

    // ---- dist = hp' + rv' - tp'; score = GAMMA - ||dist||2 -------------
    float sq = 0.f;
#pragma unroll
    for (int et = 0; et < 8; ++et) {
        float rvv = rv[et * 16 + m] * invR;
        float d   = hp8[et] * invH + rvv - tp8[et] * invT;
        float dd  = d * d;
        sq += (lane < 16) ? dd : 0.0f;   // only lanes 0-15 carry real columns
    }
#pragma unroll
    for (int mm = 16; mm >= 1; mm >>= 1) {
        sq += __shfl_xor(sq, mm, 32);
    }
    if (lane == 0) out[b] = GAMMA - sqrtf(sq);
}

// ---------------------------------------------------------------------------
extern "C" void kernel_launch(void* const* d_in, const int* in_sizes, int n_in,
                              void* d_out, int out_size, void* d_ws, size_t ws_size,
                              hipStream_t stream) {
    const int*   h_idx = (const int*)d_in[0];
    const int*   r_idx = (const int*)d_in[1];
    const int*   t_idx = (const int*)d_in[2];
    const float* ent_w = (const float*)d_in[3];
    const float* rel_w = (const float*)d_in[4];
    const float* mat_w = (const float*)d_in[5];
    float* out = (float*)d_out;

    const int B     = in_sizes[0];
    const int n_rel = in_sizes[4] / DIM;           // 500

    _Float16* mat16 = (_Float16*)d_ws;             // needs n_rel*DIM*DIM*2 B = 16 MB
    const int total = n_rel * DIM * DIM;           // 8.192M halves

    transr_repack_f16<<<(total + 255) / 256, 256, 0, stream>>>(mat_w, mat16, total);

    const int wavesPerBlock = 8;                   // 256 threads = 8 wave32
    const int blocks = (B + wavesPerBlock - 1) / wavesPerBlock;
    transr_score<<<blocks, 256, 0, stream>>>(h_idx, r_idx, t_idx,
                                             ent_w, rel_w, mat16, out, B, n_rel);
}
